// MlpLSTM_2267742732779
// MI455X (gfx1250) — matrix operand order repaired and checked by reading, not compiled, not run on hardware
//
#include <hip/hip_runtime.h>
#include <math.h>

// Problem sizes
constexpr int kB    = 256;
constexpr int kS    = 128;
constexpr int kOBS  = 256;
constexpr int kH    = 1024;
constexpr int kA    = 18;
constexpr int kG    = 4 * kH;          // 4096
constexpr int kK2   = 2 * kH;          // 2048 (packed [x ; h] K)
constexpr int kROWS = kB * kS;         // 32768

// ---------------- types ----------------
typedef __bf16        bf16x16 __attribute__((ext_vector_type(16)));
typedef float         f32x8   __attribute__((ext_vector_type(8)));
typedef unsigned int  u32x4   __attribute__((ext_vector_type(4)));

// ---------------- bf16 helpers (stored as ushort bit patterns) ----------------
__device__ __forceinline__ unsigned short f2bf(float f) {
    union { float f; unsigned u; } x; x.f = f;
    unsigned u = x.u;
    u += 0x7FFFu + ((u >> 16) & 1u);          // round-to-nearest-even
    return (unsigned short)(u >> 16);
}
__device__ __forceinline__ float bf2f(unsigned short h) {
    union { unsigned u; float f; } x; x.u = ((unsigned)h) << 16;
    return x.f;
}

// ---------------- WMMA fragment loads with immediate byte offsets ------------
// A: 16x32 (MxK): lane -> M=lane&15, K-half=(lane>>4)*16, 16 contiguous bf16
__device__ __forceinline__ bf16x16 ld_a(const unsigned short* p, int byteOfs) {
    union { u32x4 q[2]; bf16x16 v; } u;
    u.q[0] = *(const u32x4*)((const char*)p + byteOfs);
    u.q[1] = *(const u32x4*)((const char*)p + byteOfs + 16);
    return u.v;
}
// B: 32x16 (KxN), B[k,n]=W[n,k]: lane -> N=lane&15, K-half=(lane>>4)*16, 16 contiguous bf16
__device__ __forceinline__ bf16x16 ld_b(const unsigned short* p, int byteOfs) {
    union { u32x4 q[2]; bf16x16 v; } u;
    u.q[0] = *(const u32x4*)((const char*)p + byteOfs);
    u.q[1] = *(const u32x4*)((const char*)p + byteOfs + 16);
    return u.v;
}
__device__ __forceinline__ f32x8 wmma_bf16(bf16x16 a, bf16x16 b, f32x8 c) {
    return __builtin_amdgcn_wmma_f32_16x16x32_bf16(false, a, false, b, (short)0, c, false, false);
}

// ---------------- elementwise utility kernels ----------------
__global__ void cvt_f32_bf16(const float* __restrict__ src, unsigned short* __restrict__ dst, int n) {
    for (int i = blockIdx.x * blockDim.x + threadIdx.x; i < n; i += gridDim.x * blockDim.x)
        dst[i] = f2bf(src[i]);
}
__global__ void copy_f32(const float* __restrict__ src, float* __restrict__ dst, int n) {
    for (int i = blockIdx.x * blockDim.x + threadIdx.x; i < n; i += gridDim.x * blockDim.x)
        dst[i] = src[i];
}
// Pack Wc[n] = [W_ih[n] (K=0..1023) | W_hh[n] (K=1024..2047)]  (bf16)
__global__ void pack_wc(const float* __restrict__ Wih, const float* __restrict__ Whh,
                        unsigned short* __restrict__ Wc) {
    for (int i = blockIdx.x * blockDim.x + threadIdx.x; i < kG * kH; i += gridDim.x * blockDim.x) {
        int n = i / kH, k = i % kH;
        Wc[(long long)n * kK2 + k]      = f2bf(Wih[i]);
        Wc[(long long)n * kK2 + kH + k] = f2bf(Whh[i]);
    }
}
__global__ void combine_bias(const float* __restrict__ bih, const float* __restrict__ bhh,
                             float* __restrict__ bc) {
    int i = blockIdx.x * blockDim.x + threadIdx.x;
    if (i < kG) bc[i] = bih[i] + bhh[i];
}
// Ain[b] = [ x[b, s=0, :] | bf16(hx[b,:]) ]
__global__ void init_ain(const unsigned short* __restrict__ xbuf, const float* __restrict__ hx,
                         unsigned short* __restrict__ Ain) {
    int idx = blockIdx.x * blockDim.x + threadIdx.x;
    if (idx >= kB * kH) return;
    int b = idx / kH, j = idx % kH;
    Ain[(long long)b * kK2 + j]      = xbuf[((long long)b * kS) * kH + j];
    Ain[(long long)b * kK2 + kH + j] = f2bf(hx[idx]);
}

// =============================================================================
// Shared WMMA inner core: wave tile 32(M) x 64(N), K-loop unroll-2 ping-pong.
// STRIDE = K stride (elements) of both A and W. NK = K/32 (even, >= 4).
// pa/pb are per-lane induction pointers; all fragment loads use immediate
// byte offsets (A rows: i*16*STRIDE*2 ; substep: +64B) so the loop body is
// 2 pointer adds + 24 b128 loads + 16 WMMA with no register rotation.
// =============================================================================
template <int STRIDE, int NK>
__device__ __forceinline__ void wmma_core(const unsigned short* pa,
                                          const unsigned short* pb,
                                          f32x8 (&c)[2][4]) {
    constexpr int AOF = 16 * STRIDE * 2;   // bytes between A row sub-tiles
    constexpr int BOF = 16 * STRIDE * 2;   // bytes between B column sub-tiles
    bf16x16 A0[2], B0[4], A1[2], B1[4];
#pragma unroll
    for (int i = 0; i < 2; ++i) A0[i] = ld_a(pa, i * AOF);
#pragma unroll
    for (int j = 0; j < 4; ++j) B0[j] = ld_b(pb, j * BOF);

    for (int t = 0; t < NK - 2; t += 2) {
#pragma unroll
        for (int i = 0; i < 2; ++i) A1[i] = ld_a(pa, i * AOF + 64);
#pragma unroll
        for (int j = 0; j < 4; ++j) B1[j] = ld_b(pb, j * BOF + 64);
#pragma unroll
        for (int i = 0; i < 2; ++i)
#pragma unroll
            for (int j = 0; j < 4; ++j) c[i][j] = wmma_bf16(A0[i], B0[j], c[i][j]);
#pragma unroll
        for (int i = 0; i < 2; ++i) A0[i] = ld_a(pa, i * AOF + 128);
#pragma unroll
        for (int j = 0; j < 4; ++j) B0[j] = ld_b(pb, j * BOF + 128);
#pragma unroll
        for (int i = 0; i < 2; ++i)
#pragma unroll
            for (int j = 0; j < 4; ++j) c[i][j] = wmma_bf16(A1[i], B1[j], c[i][j]);
        pa += 64;   // 64 elements = 128 bytes = two K-steps
        pb += 64;
    }
    // tail: S0 holds kstep NK-2; load and process kstep NK-1
#pragma unroll
    for (int i = 0; i < 2; ++i) A1[i] = ld_a(pa, i * AOF + 64);
#pragma unroll
    for (int j = 0; j < 4; ++j) B1[j] = ld_b(pb, j * BOF + 64);
#pragma unroll
    for (int i = 0; i < 2; ++i)
#pragma unroll
        for (int j = 0; j < 4; ++j) c[i][j] = wmma_bf16(A0[i], B0[j], c[i][j]);
#pragma unroll
    for (int i = 0; i < 2; ++i)
#pragma unroll
        for (int j = 0; j < 4; ++j) c[i][j] = wmma_bf16(A1[i], B1[j], c[i][j]);
}

// ---------------- body GEMM: x = relu(obs @ W_body^T + b_body) -> bf16 -------
// M = kROWS, K = kOBS, N = kH. Block 256 thr = 8 waves (2M x 4N); wave 32x64.
__global__ __launch_bounds__(256)
void gemm_body(const unsigned short* __restrict__ obs_bf,
               const unsigned short* __restrict__ Wb,
               const float* __restrict__ bias,
               unsigned short* __restrict__ xout) {
    int wv = threadIdx.x >> 5, lane = threadIdx.x & 31;
    int m0 = blockIdx.x * 64 + (wv & 1) * 32;
    int n0 = blockIdx.y * 256 + (wv >> 1) * 64;

    const unsigned short* pa = obs_bf + (long long)(m0 + (lane & 15)) * kOBS + ((lane >> 4) * 16);
    const unsigned short* pb = Wb     + (long long)(n0 + (lane & 15)) * kOBS + ((lane >> 4) * 16);

    f32x8 c[2][4];
#pragma unroll
    for (int i = 0; i < 2; ++i)
#pragma unroll
        for (int j = 0; j < 4; ++j) c[i][j] = f32x8{0};

    wmma_core<kOBS, kOBS / 32>(pa, pb, c);

#pragma unroll
    for (int i = 0; i < 2; ++i)
#pragma unroll
        for (int j = 0; j < 4; ++j)
#pragma unroll
            for (int r = 0; r < 8; ++r) {
                int m = m0 + i * 16 + r + 8 * (lane >> 4);
                int n = n0 + j * 16 + (lane & 15);
                float v = c[i][j][r] + bias[n];
                v = v > 0.f ? v : 0.f;
                xout[(long long)m * kH + n] = f2bf(v);
            }
}

// ---------------- per-step gate GEMM:  G = Ain @ Wc^T + bc -------------------
// M = kB, N = kG, K = kK2. Block 128 thr = 4 waves (N split); wave 32x64.
__global__ __launch_bounds__(128)
void gemm_step(const unsigned short* __restrict__ Ain,
               const unsigned short* __restrict__ Wc,
               const float* __restrict__ bc,
               float* __restrict__ gbuf) {
    int wv = threadIdx.x >> 5, lane = threadIdx.x & 31;
    int m0 = blockIdx.x * 32;
    int n0 = blockIdx.y * 256 + wv * 64;

    const unsigned short* pa = Ain + (long long)(m0 + (lane & 15)) * kK2 + ((lane >> 4) * 16);
    const unsigned short* pb = Wc  + (long long)(n0 + (lane & 15)) * kK2 + ((lane >> 4) * 16);

    f32x8 c[2][4];
#pragma unroll
    for (int i = 0; i < 2; ++i)
#pragma unroll
        for (int j = 0; j < 4; ++j) c[i][j] = f32x8{0};

    wmma_core<kK2, kK2 / 32>(pa, pb, c);

#pragma unroll
    for (int i = 0; i < 2; ++i)
#pragma unroll
        for (int j = 0; j < 4; ++j)
#pragma unroll
            for (int r = 0; r < 8; ++r) {
                int m = m0 + i * 16 + r + 8 * (lane >> 4);
                int n = n0 + j * 16 + (lane & 15);
                gbuf[(long long)m * kG + n] = c[i][j][r] + bc[n];
            }
}

// ---------------- pointwise LSTM cell + staging of next step's A -------------
__global__ __launch_bounds__(256)
void lstm_point(const float* __restrict__ gbuf, float* __restrict__ cbuf,
                const unsigned short* __restrict__ xbuf,
                unsigned short* __restrict__ Ain,
                unsigned short* __restrict__ hsbuf, int s) {
    int idx = blockIdx.x * blockDim.x + threadIdx.x;   // over kB*kH
    if (idx >= kB * kH) return;
    int b = idx / kH, j = idx % kH;
    const float* g = gbuf + (long long)b * kG;
    float gi = 1.f / (1.f + expf(-g[j]));
    float gf = 1.f / (1.f + expf(-g[kH + j]));
    float gg = tanhf(g[2 * kH + j]);
    float go = 1.f / (1.f + expf(-g[3 * kH + j]));
    float c = gf * cbuf[idx] + gi * gg;
    float h = go * tanhf(c);
    cbuf[idx] = c;
    unsigned short hb = f2bf(h);
    hsbuf[((long long)b * kS + s) * kH + j] = hb;
    Ain[(long long)b * kK2 + kH + j] = hb;                     // h for next step
    if (s + 1 < kS)                                            // x_{s+1} for next step
        Ain[(long long)b * kK2 + j] = xbuf[((long long)b * kS + (s + 1)) * kH + j];
}

// ---------------- policy/value head: one wave per (b,s) row ------------------
__global__ __launch_bounds__(256)
void head_kernel(const unsigned short* __restrict__ hs,
                 const float* __restrict__ Wl, const float* __restrict__ bl,
                 const float* __restrict__ Wv, const float* __restrict__ bv,
                 const int* __restrict__ acts, float* __restrict__ out) {
    int wv = threadIdx.x >> 5, lane = threadIdx.x & 31;
    int row = blockIdx.x * 8 + wv;
    if (row >= kROWS) return;
    const unsigned short* hp = hs + (long long)row * kH;
    float acc[kA + 1];
#pragma unroll
    for (int a = 0; a <= kA; ++a) acc[a] = 0.f;
    for (int k = lane; k < kH; k += 32) {
        float hv = bf2f(hp[k]);
#pragma unroll
        for (int a = 0; a < kA; ++a) acc[a] += hv * Wl[a * kH + k];
        acc[kA] += hv * Wv[k];
    }
#pragma unroll
    for (int a = 0; a <= kA; ++a)
        for (int off = 16; off > 0; off >>= 1)
            acc[a] += __shfl_xor(acc[a], off, 32);
    if (lane == 0) {
        float logits[kA];
        float mx = -1e30f;
#pragma unroll
        for (int a = 0; a < kA; ++a) { logits[a] = acc[a] + bl[a]; mx = fmaxf(mx, logits[a]); }
        float se = 0.f;
#pragma unroll
        for (int a = 0; a < kA; ++a) se += expf(logits[a] - mx);
        float lse = mx + logf(se);
        int act = acts[row];
        float ent = 0.f, lp_act = 0.f;
        const long long O1 = (long long)kROWS * kA;
        const long long O2 = O1 + kROWS;
        const long long O3 = O2 + kROWS;
#pragma unroll
        for (int a = 0; a < kA; ++a) {
            float lp = logits[a] - lse;
            out[(long long)row * kA + a] = lp;
            ent -= expf(lp) * lp;
            if (a == act) lp_act = lp;
        }
        out[O1 + row] = lp_act;
        out[O2 + row] = ent;
        out[O3 + row] = acc[kA] + bv[0];
    }
}

// ---------------- host driver ------------------------------------------------
extern "C" void kernel_launch(void* const* d_in, const int* in_sizes, int n_in,
                              void* d_out, int out_size, void* d_ws, size_t ws_size,
                              hipStream_t stream) {
    (void)in_sizes; (void)n_in; (void)out_size; (void)ws_size;
    const float* obs    = (const float*)d_in[0];
    const float* hx     = (const float*)d_in[1];
    const float* cx     = (const float*)d_in[2];
    const int*   bacts  = (const int*)  d_in[3];
    const float* W_body = (const float*)d_in[4];
    const float* b_body = (const float*)d_in[5];
    const float* W_ih   = (const float*)d_in[6];
    const float* b_ih   = (const float*)d_in[7];
    const float* W_hh   = (const float*)d_in[8];
    const float* b_hh   = (const float*)d_in[9];
    const float* W_v    = (const float*)d_in[10];
    const float* b_v    = (const float*)d_in[11];
    const float* W_l    = (const float*)d_in[12];
    const float* b_l    = (const float*)d_in[13];
    float* out = (float*)d_out;

    // Workspace carve-up (bytes)
    char* ws = (char*)d_ws;
    size_t off = 0;
    auto carve = [&](size_t bytes) { char* p = ws + off; off += (bytes + 255) & ~(size_t)255; return p; };
    unsigned short* obs_bf = (unsigned short*)carve((size_t)kROWS * kOBS * 2); // 16 MB
    unsigned short* Wb_bf  = (unsigned short*)carve((size_t)kH * kOBS * 2);    // 0.5 MB
    unsigned short* Wc     = (unsigned short*)carve((size_t)kG * kK2 * 2);     // 16 MB
    unsigned short* xbuf   = (unsigned short*)carve((size_t)kROWS * kH * 2);   // 64 MB
    unsigned short* hsbuf  = (unsigned short*)carve((size_t)kROWS * kH * 2);   // 64 MB
    float*          gbuf   = (float*)carve((size_t)kB * kG * 4);               // 4 MB
    unsigned short* Ain    = (unsigned short*)carve((size_t)kB * kK2 * 2);     // 1 MB
    float*          cbuf   = (float*)carve((size_t)kB * kH * 4);               // 1 MB
    float*          bc     = (float*)carve((size_t)kG * 4);                    // 16 KB

    // 1) precision staging
    cvt_f32_bf16<<<512, 256, 0, stream>>>(obs,    obs_bf, kROWS * kOBS);
    cvt_f32_bf16<<<128, 256, 0, stream>>>(W_body, Wb_bf,  kH * kOBS);
    pack_wc     <<<512, 256, 0, stream>>>(W_ih, W_hh, Wc);
    combine_bias<<<kG / 256, 256, 0, stream>>>(b_ih, b_hh, bc);
    copy_f32    <<<128, 256, 0, stream>>>(cx, cbuf, kB * kH);

    // 2) body MLP (WMMA bf16)
    gemm_body<<<dim3(kROWS / 64, kH / 256), 256, 0, stream>>>(obs_bf, Wb_bf, b_body, xbuf);

    // 3) stage Ain = [x_0 | hx]
    init_ain<<<(kB * kH) / 256, 256, 0, stream>>>(xbuf, hx, Ain);

    // 4) LSTM recurrence: per step one packed-K GEMM + pointwise cell/staging
    for (int s = 0; s < kS; ++s) {
        gemm_step<<<dim3(kB / 32, kG / 256), 128, 0, stream>>>(Ain, Wc, bc, gbuf);
        lstm_point<<<(kB * kH) / 256, 256, 0, stream>>>(gbuf, cbuf, xbuf, Ain, hsbuf, s);
    }

    // 5) policy/value head
    head_kernel<<<kROWS / 8, 256, 0, stream>>>(hsbuf, W_l, b_l, W_v, b_v, bacts, out);
}
